// BasisV_filter_42296837931756
// MI455X (gfx1250) — matrix operand order repaired and checked by reading, not compiled, not run on hardware
//
#include <hip/hip_runtime.h>

typedef __attribute__((ext_vector_type(16))) _Float16 v16h;
typedef __attribute__((ext_vector_type(8)))  _Float16 v8h;
typedef __attribute__((ext_vector_type(8)))  float    v8f;
typedef __attribute__((ext_vector_type(2)))  int      v2i;

#define LEAKY(x) ((x) > 0.f ? (x) : 0.2f * (x))

#if defined(__has_builtin)
#  if __has_builtin(__builtin_amdgcn_global_load_async_to_lds_b64) && \
      __has_builtin(__builtin_amdgcn_s_wait_asynccnt)
#    define USE_ASYNC_LDS 1
#  endif
#endif
#ifndef USE_ASYNC_LDS
#  define USE_ASYNC_LDS 0
#endif

#if USE_ASYNC_LDS
// Builtin signature (from hipcc diagnostic): pointee is a 2-int vector.
// AS3 <-> flat is aperture|offset on AMDGPU, so integer truncation is exact.
#define GPTR(p) ((__attribute__((address_space(1))) v2i*)(unsigned long long)(p))
#define LPTR(p) ((__attribute__((address_space(3))) v2i*)(unsigned long long)(unsigned int)(unsigned long long)(p))
#endif

// ------------------------------------------------------------------ utilities
__global__ void k_zero(float* __restrict__ p, int n) {
  int i = blockIdx.x * 256 + threadIdx.x;
  if (i < n) p[i] = 0.f;
}

// img_f NCHW f32 -> padded NHWC f16 [16][66][66][64]
__global__ void k_prep_img(const float* __restrict__ img, _Float16* __restrict__ outp) {
  int idx = blockIdx.x * 256 + threadIdx.x;
  const int total = 16 * 66 * 66 * 64;
  if (idx >= total) return;
  int c = idx & 63; int t = idx >> 6;
  int xp = t % 66; t /= 66;
  int yp = t % 66; int n = t / 66;
  float v = 0.f;
  if (xp >= 1 && xp <= 64 && yp >= 1 && yp <= 64)
    v = img[(((size_t)(n * 64 + c)) * 64 + (yp - 1)) * 64 + (xp - 1)];
  outp[idx] = (_Float16)v;
}

// OIHW f32 -> Bt[oc][k] f16, k = (ky*3+kx)*IC + ic
template <int IC, int OC>
__global__ void k_prep_w(const float* __restrict__ wsrc, _Float16* __restrict__ Bt) {
  int idx = blockIdx.x * 256 + threadIdx.x;
  const int K = IC * 9;
  if (idx >= OC * K) return;
  int k = idx % K; int oc = idx / K;
  int ic = k % IC; int pos = k / IC;
  int ky = pos / 3, kx = pos % 3;
  Bt[idx] = (_Float16)wsrc[(((size_t)(oc * IC + ic)) * 3 + ky) * 3 + kx];
}

// --------------------------------------------------------- WMMA implicit GEMM
// inpad: padded NHWC f16 [16][66][66][IC];  Bt: [OC][IC*9] f16
// rawout: NHWC f16 [65536][OC];  stats: per-channel sum / sumsq (f32 atomics)
// Block: 256 threads = 8 waves. Block tile M=256 pixels x N=32 oc.
// Wave tile: 2 M-subtiles x 2 N-subtiles = 4 WMMAs/chunk.
// B chunk (32 oc x 32 k) staged in LDS, double-buffered via async DMA.
template <int IC, int OC>
__launch_bounds__(256)
__global__ void k_conv_wmma(const _Float16* __restrict__ inpad,
                            const _Float16* __restrict__ Bt,
                            _Float16* __restrict__ rawout,
                            float* __restrict__ stat_sum,
                            float* __restrict__ stat_sq) {
  constexpr int K   = IC * 9;
  constexpr int KC  = K / 32;    // 32-wide K chunks
  constexpr int ICC = IC / 32;   // chunks per 3x3 kernel position
  __shared__ float ssum[32], ssq[32];
  __shared__ _Float16 sB[2][32 * 32];  // [buf][oc_local][k_in_chunk]
  if (threadIdx.x < 32) { ssum[threadIdx.x] = 0.f; ssq[threadIdx.x] = 0.f; }

  const int lane = threadIdx.x & 31;
  const int wave = threadIdx.x >> 5;              // 8 waves
  const int p0   = blockIdx.x * 256 + wave * 32;  // 32 consecutive pixels per wave
  const int oc0  = blockIdx.y * 32;
  const int mA   = lane & 15;                     // A row / B column / D column
  const int kh   = lane >> 4;                     // K-half select
  const int n    = p0 >> 12;
  const int yy   = (p0 >> 6) & 63;
  const int x0   = (p0 & 63) + mA;                // subtile 0 column
  const int x1   = x0 + 16;                       // subtile 1 column

  // staging: thread moves 8 bytes of the 2KB B chunk
  const int sj = threadIdx.x >> 3;                // oc row 0..31
  const int sq = threadIdx.x & 7;                 // 8-byte piece 0..7
  const _Float16* gB = Bt + (size_t)(oc0 + sj) * K + sq * 4;
  _Float16* lB0 = &sB[0][sj * 32 + sq * 4];
  _Float16* lB1 = &sB[1][sj * 32 + sq * 4];

#define STAGE_B(bufp, kc)                                                        \
  do {                                                                           \
    const _Float16* gp_ = gB + (kc) * 32;                                        \
    _Float16* lp_ = (bufp) ? lB1 : lB0;                                          \
    STAGE_COPY(gp_, lp_);                                                        \
  } while (0)
#if USE_ASYNC_LDS
#define STAGE_COPY(gp_, lp_) \
  __builtin_amdgcn_global_load_async_to_lds_b64(GPTR(gp_), LPTR(lp_), 0, 0)
#else
#define STAGE_COPY(gp_, lp_) \
  *(unsigned long long*)(lp_) = *(const unsigned long long*)(gp_)
#endif

  __builtin_prefetch(inpad + (((size_t)(n * 66 + yy)) * 66 + x0) * IC, 0, 0);
  __builtin_prefetch(inpad + (((size_t)(n * 66 + yy)) * 66 + x1) * IC, 0, 0);

  STAGE_B(0, 0);

  v8f acc00 = {}, acc01 = {}, acc10 = {}, acc11 = {};
  int buf = 0;
  for (int kc = 0; kc < KC; ++kc) {
#if USE_ASYNC_LDS
    __builtin_amdgcn_s_wait_asynccnt(0);   // own stage of sB[buf] landed
#endif
    __syncthreads();                       // all waves staged; prev reads done
    if (kc + 1 < KC) STAGE_B(buf ^ 1, kc + 1);  // DMA next chunk during compute

    int pos = kc / ICC;
    int icb = (kc % ICC) * 32;
    int ky = pos / 3, kx = pos - ky * 3;
    // A fragments: lanes<16 hold K {0..15}; lanes>=16 hold {16..31}
    const _Float16* ap0 =
        inpad + (((size_t)(n * 66 + (yy + ky))) * 66 + (x0 + kx)) * IC + icb + kh * 16;
    const _Float16* ap1 =
        inpad + (((size_t)(n * 66 + (yy + ky))) * 66 + (x1 + kx)) * IC + icb + kh * 16;
    union { v16h v; v8h h[2]; } a0, a1;
    a0.h[0] = *(const v8h*)ap0; a0.h[1] = *(const v8h*)(ap0 + 8);
    a1.h[0] = *(const v8h*)ap1; a1.h[1] = *(const v8h*)(ap1 + 8);
    // B fragments from LDS: lane holds column N, K-range kh*16..+15
    const _Float16* bl = &sB[buf][0];
    v16h b0 = *(const v16h*)(bl + mA * 32 + kh * 16);
    v16h b1 = *(const v16h*)(bl + (16 + mA) * 32 + kh * 16);

    acc00 = __builtin_amdgcn_wmma_f32_16x16x32_f16(false, a0.v, false, b0, (short)0, acc00, false, false);
    acc01 = __builtin_amdgcn_wmma_f32_16x16x32_f16(false, a0.v, false, b1, (short)0, acc01, false, false);
    acc10 = __builtin_amdgcn_wmma_f32_16x16x32_f16(false, a1.v, false, b0, (short)0, acc10, false, false);
    acc11 = __builtin_amdgcn_wmma_f32_16x16x32_f16(false, a1.v, false, b1, (short)0, acc11, false, false);
    buf ^= 1;
  }
#undef STAGE_B
#undef STAGE_COPY

  // D layout: vgpr r -> M = r + 8*kh, N = mA
#pragma unroll
  for (int r = 0; r < 8; ++r) {
    int pD0 = p0 + r + 8 * kh;
    int pD1 = p0 + 16 + r + 8 * kh;
    float v00 = acc00[r], v01 = acc01[r], v10 = acc10[r], v11 = acc11[r];
    rawout[(size_t)pD0 * OC + oc0 + mA]      = (_Float16)v00;
    rawout[(size_t)pD0 * OC + oc0 + 16 + mA] = (_Float16)v01;
    rawout[(size_t)pD1 * OC + oc0 + mA]      = (_Float16)v10;
    rawout[(size_t)pD1 * OC + oc0 + 16 + mA] = (_Float16)v11;
    atomicAdd(&ssum[mA],      v00 + v10); atomicAdd(&ssq[mA],      v00 * v00 + v10 * v10);
    atomicAdd(&ssum[16 + mA], v01 + v11); atomicAdd(&ssq[16 + mA], v01 * v01 + v11 * v11);
  }
  __syncthreads();
  if (threadIdx.x < 32) {
    atomicAdd(&stat_sum[oc0 + threadIdx.x], ssum[threadIdx.x]);
    atomicAdd(&stat_sq [oc0 + threadIdx.x], ssq [threadIdx.x]);
  }
}

// BN1(train stats) + leaky -> padded NHWC f16 [16][66][66][128]
__global__ void k_bn1_apply(const _Float16* __restrict__ raw1, const float* __restrict__ s1,
                            const float* __restrict__ q1, const float* __restrict__ g,
                            const float* __restrict__ bb, _Float16* __restrict__ outp) {
  int idx = blockIdx.x * 256 + threadIdx.x;
  const int total = 16 * 66 * 66 * 128;
  if (idx >= total) return;
  int c = idx & 127; int t = idx >> 7;
  int xp = t % 66; t /= 66;
  int yp = t % 66; int n = t / 66;
  float v = 0.f;
  if (xp >= 1 && xp <= 64 && yp >= 1 && yp <= 64) {
    float m   = s1[c] * (1.f / 65536.f);
    float var = q1[c] * (1.f / 65536.f) - m * m;
    float inv = rsqrtf(var + 1e-5f);
    float xv = (float)raw1[(((size_t)(n * 64 + (yp - 1))) * 64 + (xp - 1)) * 128 + c];
    xv = (xv - m) * inv * g[c] + bb[c];
    v = LEAKY(xv);
  }
  outp[idx] = (_Float16)v;
}

// BN2 + leaky + global average pool -> img_emb [16][256]
__global__ void k_bn2_pool(const _Float16* __restrict__ raw2, const float* __restrict__ s2,
                           const float* __restrict__ q2, const float* __restrict__ g,
                           const float* __restrict__ bb, float* __restrict__ iemb) {
  int b = blockIdx.x, ocg = blockIdx.y;
  int ocl = threadIdx.x & 15, lp = threadIdx.x >> 4;
  int oc = ocg * 16 + ocl;
  float m   = s2[oc] * (1.f / 65536.f);
  float var = q2[oc] * (1.f / 65536.f) - m * m;
  float inv = rsqrtf(var + 1e-5f);
  float gg = g[oc], bv = bb[oc];
  float s = 0.f;
  for (int p = lp; p < 4096; p += 16) {
    float xv = (float)raw2[((size_t)(b * 4096 + p)) * 256 + oc];
    xv = (xv - m) * inv * gg + bv;
    s += LEAKY(xv);
  }
  __shared__ float red[256];
  red[threadIdx.x] = s; __syncthreads();
  for (int st = 128; st >= 16; st >>= 1) {
    if (threadIdx.x < st) red[threadIdx.x] += red[threadIdx.x + st];
    __syncthreads();
  }
  if (threadIdx.x < 16) iemb[b * 256 + ocg * 16 + threadIdx.x] = red[threadIdx.x] * (1.f / 4096.f);
}

// ------------------------------------------------------------ task branch
__global__ void k_task_rms(const float* __restrict__ tf, float* __restrict__ invr) {
  int b = blockIdx.x;
  const float* xp = tf + (size_t)b * 16384;
  float s = 0.f;
  for (int i = threadIdx.x; i < 16384; i += 256) { float v = xp[i]; s += v * v; }
  __shared__ float red[256];
  red[threadIdx.x] = s; __syncthreads();
  for (int st = 128; st > 0; st >>= 1) {
    if (threadIdx.x < st) red[threadIdx.x] += red[threadIdx.x + st];
    __syncthreads();
  }
  if (threadIdx.x == 0) {
    float rms = sqrtf(red[0] * (1.f / 16384.f));
    invr[b] = 1.f / (rms + 1e-8f);
  }
}

// per (b,c): S = sum_t leaky(conv1d(x)), plus endpoints h[0], h[L-1]
__global__ void k_task_stats(const float* __restrict__ tf, const float* __restrict__ invr,
                             const float* __restrict__ c1w, float* __restrict__ tS,
                             float* __restrict__ tH0, float* __restrict__ tHL) {
  int c = blockIdx.x, b = blockIdx.y;
  float w0 = c1w[c * 3], w1 = c1w[c * 3 + 1], w2 = c1w[c * 3 + 2];
  const float* xp = tf + (size_t)b * 16384;
  float ir = invr[b];
  float s = 0.f;
  for (int t = threadIdx.x; t < 16384; t += 256) {
    float xm = (t > 0)     ? xp[t - 1] * ir : 0.f;
    float x0 = xp[t] * ir;
    float xq = (t < 16383) ? xp[t + 1] * ir : 0.f;
    float h = w0 * xm + w1 * x0 + w2 * xq;
    h = LEAKY(h);
    s += h;
    if (t == 0)     tH0[b * 128 + c] = h;
    if (t == 16383) tHL[b * 128 + c] = h;
  }
  __shared__ float red[256];
  red[threadIdx.x] = s; __syncthreads();
  for (int st = 128; st > 0; st >>= 1) {
    if (threadIdx.x < st) red[threadIdx.x] += red[threadIdx.x + st];
    __syncthreads();
  }
  if (threadIdx.x == 0) tS[b * 128 + c] = red[0];
}

// task_emb[b][o] = mean_t conv1d(h1, c2w) collapsed to endpoint-corrected sums
__global__ void k_task_emb(const float* __restrict__ tS, const float* __restrict__ tH0,
                           const float* __restrict__ tHL, const float* __restrict__ c2w,
                           float* __restrict__ temb) {
  int b = blockIdx.x, o = threadIdx.x;  // 256 threads
  __shared__ float S[128], H0[128], HL[128];
  if (o < 128) { S[o] = tS[b * 128 + o]; H0[o] = tH0[b * 128 + o]; HL[o] = tHL[b * 128 + o]; }
  __syncthreads();
  float acc = 0.f;
  for (int i = 0; i < 128; ++i) {
    const float* w = c2w + ((size_t)o * 128 + i) * 3;
    float Si = S[i];
    acc += w[0] * (Si - HL[i]) + w[1] * Si + w[2] * (Si - H0[i]);
  }
  temb[b * 256 + o] = acc * (1.f / 16384.f);
}

// ------------------------------------------------------------ MLP + LayerNorm
__global__ void k_mlp(const float* __restrict__ temb, const float* __restrict__ iemb,
                      const float* __restrict__ w1, const float* __restrict__ b1,
                      const float* __restrict__ lg, const float* __restrict__ lb,
                      const float* __restrict__ w2, const float* __restrict__ b2,
                      float* __restrict__ filt, float* __restrict__ invfn) {
  int b = blockIdx.x, j = threadIdx.x;  // 128 threads
  __shared__ float emb[512], hh[128], red[128];
  for (int i = j; i < 256; i += 128) { emb[i] = temb[b * 256 + i]; emb[256 + i] = iemb[b * 256 + i]; }
  __syncthreads();
  float h = b1[j];
  for (int d = 0; d < 512; ++d) h += emb[d] * w1[d * 128 + j];
  red[j] = h; __syncthreads();
  for (int st = 64; st > 0; st >>= 1) { if (j < st) red[j] += red[j + st]; __syncthreads(); }
  float mu = red[0] * (1.f / 128.f);
  __syncthreads();
  float dv = h - mu;
  red[j] = dv * dv; __syncthreads();
  for (int st = 64; st > 0; st >>= 1) { if (j < st) red[j] += red[j + st]; __syncthreads(); }
  float var = red[0] * (1.f / 128.f);
  float hn = dv * rsqrtf(var + 1e-5f) * lg[j] + lb[j];
  hh[j] = hn > 0.f ? hn : 0.f;
  __syncthreads();
  float local = 0.f;
  for (int q = 0; q < 4; ++q) {
    int o = j * 4 + q;
    float a = b2[o];
    for (int i = 0; i < 128; ++i) a += hh[i] * w2[i * 512 + o];
    filt[b * 512 + o] = a;
    local += a * a;
  }
  red[j] = local; __syncthreads();
  for (int st = 64; st > 0; st >>= 1) { if (j < st) red[j] += red[j + st]; __syncthreads(); }
  if (j == 0) invfn[b] = 1.f / fmaxf(sqrtf(red[0]), 1e-12f);
}

// -------------------------------------------- cosine-rank selection + gather
__global__ void k_select(const float* __restrict__ bank, const int* __restrict__ labels,
                         const float* __restrict__ filt, const float* __restrict__ invfn,
                         float* __restrict__ out) {
  int f = blockIdx.x, b = blockIdx.y, v = threadIdx.x;  // 64 threads
  __shared__ int lab[32];
  __shared__ float sf[512];
  __shared__ float sims[64];
  __shared__ int sfirst, ssel;
  if (v < 32) lab[v] = labels[b * 32 + v];
  for (int i = v; i < 512; i += 64) sf[i] = filt[b * 512 + i];
  __syncthreads();
  if (v == 0) { int lf = lab[f]; int fi = 0; while (lab[fi] != lf) fi++; sfirst = fi; }
  __syncthreads();
  int first = sfirst;
  int rank = f - first;
  const float* row = bank + (((size_t)(b * 32 + first)) * 64 + v) * 512;
  float dot = 0.f, nn = 0.f;
  for (int d = 0; d < 512; ++d) { float r = row[d]; dot += sf[d] * r; nn += r * r; }
  sims[v] = dot * invfn[b] / fmaxf(sqrtf(nn), 1e-12f);
  __syncthreads();
  float sv = sims[v];
  int pos = 0;
  for (int u = 0; u < 64; ++u) {
    float su = sims[u];
    pos += (su > sv) || (su == sv && u < v);  // stable descending order position
  }
  if (pos == rank) ssel = v;
  __syncthreads();
  const float* srow = bank + (((size_t)(b * 32 + first)) * 64 + ssel) * 512;
  float* op = out + ((size_t)(b * 32 + f)) * 512;
  for (int i = v; i < 512; i += 64) op[i] = srow[i];
}

// ------------------------------------------------------------------- launcher
static inline size_t alignUp(size_t x, size_t a) { return (x + a - 1) & ~(a - 1); }

extern "C" void kernel_launch(void* const* d_in, const int* in_sizes, int n_in,
                              void* d_out, int out_size, void* d_ws, size_t ws_size,
                              hipStream_t stream) {
  (void)in_sizes; (void)n_in; (void)out_size; (void)ws_size;
  const float* bank   = (const float*)d_in[0];
  const float* task_f = (const float*)d_in[1];
  const float* img_f  = (const float*)d_in[2];
  const int*   labels = (const int*)d_in[3];
  const float* c1w    = (const float*)d_in[4];
  const float* c2w    = (const float*)d_in[5];
  const float* i1w    = (const float*)d_in[6];
  const float* bn1g   = (const float*)d_in[7];
  const float* bn1b   = (const float*)d_in[8];
  const float* i2w    = (const float*)d_in[9];
  const float* bn2g   = (const float*)d_in[10];
  const float* bn2b   = (const float*)d_in[11];
  const float* w1     = (const float*)d_in[12];
  const float* b1     = (const float*)d_in[13];
  const float* lng    = (const float*)d_in[14];
  const float* lnb    = (const float*)d_in[15];
  const float* w2     = (const float*)d_in[16];
  const float* b2     = (const float*)d_in[17];
  float* out = (float*)d_out;

  char* wsb = (char*)d_ws;
  size_t off = 0;
  auto take = [&](size_t bytes) -> void* { void* p = wsb + off; off = alignUp(off + bytes, 256); return p; };

  _Float16* imgh = (_Float16*)take((size_t)16 * 66 * 66 * 64 * 2);
  _Float16* Bt1  = (_Float16*)take((size_t)128 * 576 * 2);
  _Float16* Bt2  = (_Float16*)take((size_t)256 * 1152 * 2);
  _Float16* raw1 = (_Float16*)take((size_t)16 * 64 * 64 * 128 * 2);
  _Float16* a1   = (_Float16*)take((size_t)16 * 66 * 66 * 128 * 2);
  _Float16* raw2 = (_Float16*)take((size_t)16 * 64 * 64 * 256 * 2);
  float* stats = (float*)take(768 * 4);  // s1[128] q1[128] s2[256] q2[256]
  float* s1 = stats, *q1 = stats + 128, *s2 = stats + 256, *q2 = stats + 512;
  float* invr  = (float*)take(16 * 4);
  float* tS    = (float*)take(2048 * 4);
  float* tH0   = (float*)take(2048 * 4);
  float* tHL   = (float*)take(2048 * 4);
  float* temb  = (float*)take(4096 * 4);
  float* iemb  = (float*)take(4096 * 4);
  float* filt  = (float*)take(8192 * 4);
  float* invfn = (float*)take(16 * 4);

  k_zero<<<3, 256, 0, stream>>>(stats, 768);
  k_prep_img<<<(16 * 66 * 66 * 64) / 256, 256, 0, stream>>>(img_f, imgh);
  k_prep_w<64, 128><<<(128 * 576) / 256, 256, 0, stream>>>(i1w, Bt1);
  k_prep_w<128, 256><<<(256 * 1152) / 256, 256, 0, stream>>>(i2w, Bt2);

  k_conv_wmma<64, 128><<<dim3(256, 4), 256, 0, stream>>>(imgh, Bt1, raw1, s1, q1);
  k_bn1_apply<<<(16 * 66 * 66 * 128) / 256, 256, 0, stream>>>(raw1, s1, q1, bn1g, bn1b, a1);
  k_conv_wmma<128, 256><<<dim3(256, 8), 256, 0, stream>>>(a1, Bt2, raw2, s2, q2);
  k_bn2_pool<<<dim3(16, 16), 256, 0, stream>>>(raw2, s2, q2, bn2g, bn2b, iemb);

  k_task_rms<<<16, 256, 0, stream>>>(task_f, invr);
  k_task_stats<<<dim3(128, 16), 256, 0, stream>>>(task_f, invr, c1w, tS, tH0, tHL);
  k_task_emb<<<16, 256, 0, stream>>>(tS, tH0, tHL, c2w, temb);

  k_mlp<<<16, 128, 0, stream>>>(temb, iemb, w1, b1, lng, lnb, w2, b2, filt, invfn);
  k_select<<<dim3(32, 16), 64, 0, stream>>>(bank, labels, filt, invfn, out);
}
